// VectorQuantizer_28698971472437
// MI455X (gfx1250) — matrix operand order repaired and checked by reading, not compiled, not run on hardware
//
#include <hip/hip_runtime.h>

// ---------- types ----------
typedef __attribute__((ext_vector_type(16))) __bf16          v16bf;
typedef __attribute__((ext_vector_type(8)))  __bf16          v8bf;
typedef __attribute__((ext_vector_type(8)))  float           v8f;
typedef __attribute__((ext_vector_type(4)))  float           v4f;
typedef __attribute__((ext_vector_type(8)))  unsigned short  v8u;
typedef __attribute__((ext_vector_type(4)))  unsigned short  v4u;
typedef int vq_i4 __attribute__((__vector_size__(16)));      // matches async-LDS builtin param

#define CAT16(a, b) __builtin_shufflevector((a), (b), 0,1,2,3,4,5,6,7,8,9,10,11,12,13,14,15)

// problem sizes (fixed by the reference)
#define VQ_N 32768
#define VQ_K 4096
#define VQ_D 256
#define LDSROW 264   // 256 bf16 + 8 pad -> 528B row stride, 132 dw == 4 mod 64 banks

// CDNA5 async global->LDS staging (ASYNCcnt path); falls back to sync loads.
#if defined(__HIP_DEVICE_COMPILE__) && __has_builtin(__builtin_amdgcn_global_load_async_to_lds_b128)
#define VQ_ASYNC 1
#else
#define VQ_ASYNC 0
#endif
#define VQ_AS1 __attribute__((address_space(1)))
#define VQ_AS3 __attribute__((address_space(3)))

template <int N>
__device__ __forceinline__ void vq_wait_async() {
#if VQ_ASYNC
#if __has_builtin(__builtin_amdgcn_s_wait_asynccnt)
    __builtin_amdgcn_s_wait_asynccnt(N);
#else
    asm volatile("s_wait_asynccnt %0" :: "i"(N) : "memory");
#endif
#endif
}

// ---------- split fp32 -> bf16 hi/lo ----------
__device__ __forceinline__ void bf_split(float f, unsigned short& h, unsigned short& l) {
    __bf16 hb = (__bf16)f;
    float  hf = (float)hb;
    __bf16 lb = (__bf16)(f - hf);
    h = __builtin_bit_cast(unsigned short, hb);
    l = __builtin_bit_cast(unsigned short, lb);
}

// ---------- kernel 1: split x into bf16 hi/lo ----------
__global__ __launch_bounds__(256) void vq_split_x(const float* __restrict__ x,
                                                  unsigned short* __restrict__ xhi,
                                                  unsigned short* __restrict__ xlo) {
    size_t i = ((size_t)blockIdx.x * 256 + threadIdx.x) * 4;   // N*D/4 threads
    v4f v = *(const v4f*)(x + i);
    v4u h, l;
#pragma unroll
    for (int j = 0; j < 4; ++j) {
        unsigned short hh, ll;
        bf_split(v[j], hh, ll);
        h[j] = hh; l[j] = ll;
    }
    *(v4u*)(xhi + i) = h;
    *(v4u*)(xlo + i) = l;
}

// ---------- kernel 2: split W + compute C2 = ||c_k||^2 ----------
__global__ __launch_bounds__(256) void vq_split_w(const float* __restrict__ w,
                                                  unsigned short* __restrict__ whi,
                                                  unsigned short* __restrict__ wlo,
                                                  float* __restrict__ c2) {
    const int t    = threadIdx.x;
    const int wave = t >> 5;
    const int lane = t & 31;
    const int row  = blockIdx.x * 8 + wave;                    // K/8 blocks, 1 wave per row
    const float* src = w + (size_t)row * VQ_D + lane * 8;
    v8u h, l;
    float ss = 0.0f;
#pragma unroll
    for (int j = 0; j < 8; ++j) {
        float f = src[j];
        ss += f * f;
        unsigned short hh, ll;
        bf_split(f, hh, ll);
        h[j] = hh; l[j] = ll;
    }
    *(v8u*)(whi + (size_t)row * VQ_D + lane * 8) = h;
    *(v8u*)(wlo + (size_t)row * VQ_D + lane * 8) = l;
#pragma unroll
    for (int off = 16; off > 0; off >>= 1) ss += __shfl_xor(ss, off, 32);
    if (lane == 0) c2[row] = ss;
}

// ---------- W-tile staging: 16 codes x 256 depth, hi+lo, 4 x 16B chunks per thread ----------
__device__ __forceinline__ void stage_tile(const unsigned short* __restrict__ Whi,
                                           const unsigned short* __restrict__ Wlo,
                                           unsigned short* lwh, unsigned short* lwl,
                                           int kt, int t) {
    const unsigned short* srcH = Whi + (size_t)kt * 16 * VQ_D;
    const unsigned short* srcL = Wlo + (size_t)kt * 16 * VQ_D;
#pragma unroll
    for (int c = 0; c < 2; ++c) {
        int chunk = t + c * 256;          // 512 chunks of 8 bf16 per matrix
        int r     = chunk >> 5;           // code row in tile
        int off   = (chunk & 31) * 8;     // depth offset
#if VQ_ASYNC
        __builtin_amdgcn_global_load_async_to_lds_b128(
            (VQ_AS1 vq_i4*)(srcH + (size_t)r * VQ_D + off),
            (VQ_AS3 vq_i4*)(&lwh[r * LDSROW + off]), 0, 0);
        __builtin_amdgcn_global_load_async_to_lds_b128(
            (VQ_AS1 vq_i4*)(srcL + (size_t)r * VQ_D + off),
            (VQ_AS3 vq_i4*)(&lwl[r * LDSROW + off]), 0, 0);
#else
        *(v8u*)(&lwh[r * LDSROW + off]) = *(const v8u*)(srcH + (size_t)r * VQ_D + off);
        *(v8u*)(&lwl[r * LDSROW + off]) = *(const v8u*)(srcL + (size_t)r * VQ_D + off);
#endif
    }
}

// ---------- kernel 3: fused split-bf16 WMMA GEMM + row argmin ----------
// block = 256 threads = 8 waves; wave owns 16 x-rows; k-loop over 256 tiles of 16 codes;
// W tile double-buffered in LDS, staged with async global->LDS copies (ASYNCcnt).
__global__ __launch_bounds__(256) void vq_argmin(const unsigned short* __restrict__ Xhi,
                                                 const unsigned short* __restrict__ Xlo,
                                                 const unsigned short* __restrict__ Whi,
                                                 const unsigned short* __restrict__ Wlo,
                                                 const float* __restrict__ C2,
                                                 int* __restrict__ closest) {
    __shared__ float lds_c2[VQ_K];
    __shared__ __attribute__((aligned(16))) unsigned short lds_wh[2][16 * LDSROW];
    __shared__ __attribute__((aligned(16))) unsigned short lds_wl[2][16 * LDSROW];

    const int t    = threadIdx.x;
    const int wave = t >> 5;
    const int lane = t & 31;
    const int hs   = lane >> 4;   // half-wave select
    const int l15  = lane & 15;
    const int rowBase = blockIdx.x * 128 + wave * 16;

#if VQ_ASYNC
    stage_tile(Whi, Wlo, lds_wh[0], lds_wl[0], 0, t);   // prologue: tile 0 in flight
#endif

    for (int i = t; i < VQ_K; i += 256) lds_c2[i] = C2[i];

    // A fragments (16x32 bf16 each): lane l holds row l&15, chunks at K=(l>>4)*8 and +16.
    v16bf a_hi[8], a_lo[8];
    {
        const unsigned short* xh = Xhi + (size_t)(rowBase + l15) * VQ_D + hs * 8;
        const unsigned short* xl = Xlo + (size_t)(rowBase + l15) * VQ_D + hs * 8;
#pragma unroll
        for (int s = 0; s < 8; ++s) {
            v8bf h0 = *(const v8bf*)(xh + s * 32);
            v8bf h1 = *(const v8bf*)(xh + s * 32 + 16);
            a_hi[s] = CAT16(h0, h1);
            v8bf l0 = *(const v8bf*)(xl + s * 32);
            v8bf l1 = *(const v8bf*)(xl + s * 32 + 16);
            a_lo[s] = CAT16(l0, l1);
        }
    }

    float bestv[8];
    int   besti[8];
#pragma unroll
    for (int m = 0; m < 8; ++m) { bestv[m] = __builtin_inff(); besti[m] = 0; }

    for (int kt = 0; kt < VQ_K / 16; ++kt) {
        const int sel = kt & 1;
        // barrier 1: nobody still reads the buffer we are about to overwrite
        __syncthreads();
#if VQ_ASYNC
        if (kt + 1 < VQ_K / 16) {
            stage_tile(Whi, Wlo, lds_wh[sel ^ 1], lds_wl[sel ^ 1], kt + 1, t);
            vq_wait_async<4>();   // 4 newest ops = next tile; current tile complete (in-order)
        } else {
            vq_wait_async<0>();
        }
#else
        stage_tile(Whi, Wlo, lds_wh[sel], lds_wl[sel], kt, t);
#endif
        // barrier 2: current tile visible to all waves
        __syncthreads();

        v8f acc0 = {}, acc1 = {}, acc2 = {};
#pragma unroll
        for (int s = 0; s < 8; ++s) {
            // B fragment (32x16): lane l holds code col l&15, chunks at K=(l>>4)*8 and +16
            const unsigned short* bh = &lds_wh[sel][l15 * LDSROW + s * 32 + hs * 8];
            const unsigned short* bl = &lds_wl[sel][l15 * LDSROW + s * 32 + hs * 8];
            v8bf b0 = *(const v8bf*)(bh);
            v8bf b1 = *(const v8bf*)(bh + 16);
            v16bf Bh = CAT16(b0, b1);
            v8bf c0 = *(const v8bf*)(bl);
            v8bf c1 = *(const v8bf*)(bl + 16);
            v16bf Bl = CAT16(c0, c1);
            // split-float product: x.w ~= xh.wh + xh.wl + xl.wh  (3 independent acc chains)
            acc0 = __builtin_amdgcn_wmma_f32_16x16x32_bf16(false, a_hi[s], false, Bh,
                                                           (short)0, acc0, false, false);
            acc1 = __builtin_amdgcn_wmma_f32_16x16x32_bf16(false, a_hi[s], false, Bl,
                                                           (short)0, acc1, false, false);
            acc2 = __builtin_amdgcn_wmma_f32_16x16x32_bf16(false, a_lo[s], false, Bh,
                                                           (short)0, acc2, false, false);
        }

        // epilogue: dist = ||c||^2 - 2*dot (||x||^2 constant per row, dropped)
        const int   col = kt * 16 + l15;
        const float c2v = lds_c2[col];
#pragma unroll
        for (int m = 0; m < 8; ++m) {
            float dist = fmaf(-2.0f, acc0[m] + acc1[m] + acc2[m], c2v);
            if (dist < bestv[m]) { bestv[m] = dist; besti[m] = col; }  // strict < keeps lowest col
        }
    }

    // cross-lane min-argmin within each 16-lane half (rows m / m+8), lowest-index tie-break
#pragma unroll
    for (int off = 1; off < 16; off <<= 1) {
#pragma unroll
        for (int m = 0; m < 8; ++m) {
            float ov = __shfl_xor(bestv[m], off, 32);
            int   oi = __shfl_xor(besti[m], off, 32);
            if (ov < bestv[m] || (ov == bestv[m] && oi < besti[m])) {
                bestv[m] = ov; besti[m] = oi;
            }
        }
    }
    if (l15 == 0) {
#pragma unroll
        for (int m = 0; m < 8; ++m)
            closest[rowBase + hs * 8 + m] = besti[m];
    }
}

// ---------- kernel 4: gather matched codebook rows (fp32) ----------
__global__ __launch_bounds__(256) void vq_gather(const float* __restrict__ w,
                                                 const int* __restrict__ closest,
                                                 float* __restrict__ out) {
    const int row  = blockIdx.x * 4 + (threadIdx.x >> 6);   // 4 rows/block, 64 lanes/row
    const int lane = threadIdx.x & 63;
    const int idx  = closest[row];
    *(v4f*)(out + (size_t)row * VQ_D + lane * 4) =
        *(const v4f*)(w + (size_t)idx * VQ_D + lane * 4);
}

// ---------- launcher ----------
extern "C" void kernel_launch(void* const* d_in, const int* in_sizes, int n_in,
                              void* d_out, int out_size, void* d_ws, size_t ws_size,
                              hipStream_t stream) {
    const float* x = (const float*)d_in[0];   // [N, D] fp32
    const float* w = (const float*)d_in[1];   // [K, D] fp32
    float* out = (float*)d_out;               // [N, D] fp32

    // workspace layout (all 256B-aligned by construction)
    const size_t XB = (size_t)VQ_N * VQ_D * 2;   // 16 MB each
    const size_t WB = (size_t)VQ_K * VQ_D * 2;   //  2 MB each
    char* ws = (char*)d_ws;
    unsigned short* Xhi = (unsigned short*)(ws);
    unsigned short* Xlo = (unsigned short*)(ws + XB);
    unsigned short* Whi = (unsigned short*)(ws + 2 * XB);
    unsigned short* Wlo = (unsigned short*)(ws + 2 * XB + WB);
    float*          C2  = (float*)(ws + 2 * XB + 2 * WB);
    int*            cls = (int*)(ws + 2 * XB + 2 * WB + (size_t)VQ_K * 4);

    vq_split_x<<<(VQ_N * VQ_D) / (256 * 4), 256, 0, stream>>>(x, Xhi, Xlo);
    vq_split_w<<<VQ_K / 8, 256, 0, stream>>>(w, Whi, Wlo, C2);
    vq_argmin<<<VQ_N / 128, 256, 0, stream>>>(Xhi, Xlo, Whi, Wlo, C2, cls);
    vq_gather<<<VQ_N / 4, 256, 0, stream>>>(w, cls, out);
}
